// LSTMModel12_21036749816063
// MI455X (gfx1250) — matrix operand order repaired and checked by reading, not compiled, not run on hardware
//
#include <hip/hip_runtime.h>
#include <hip/hip_bf16.h>

typedef __attribute__((ext_vector_type(16))) _Float16 v16h;
typedef __attribute__((ext_vector_type(8)))  float    v8f;

#define B_TOT 256
#define T_LEN 512

__device__ __forceinline__ unsigned short f2h(float f) {
  _Float16 h = (_Float16)f;
  union { _Float16 h; unsigned short u; } cvt; cvt.h = h;
  return cvt.u;
}
__device__ __forceinline__ float sigmoidf(float x) {
  return 1.0f / (1.0f + __expf(-x));
}

// One workgroup owns 16 batch rows and runs the full T-step scan for one layer.
//   - Concat weights [Wk;Wr] ((DIN+U) x 4U) swizzled once into LDS in WMMA
//     B-fragment layout (f16).
//   - Double-buffered A staging in LDS holds [x_t | h_{t-1}] (16 x (DIN+U)) in
//     WMMA A-fragment layout (f16): step t reads buf[t&1], writes h(t+1)/x(t+1)
//     into buf[(t+1)&1]  ->  a single __syncthreads() per timestep.
//   - Each wave owns one 16-unit tile of U and computes its 4 gate tiles
//     (i,f,g,o) with v_wmma_f32_16x16x32_f16; cell state c stays in VGPRs.
//   - First k-tile uses a loop-invariant zero C; bias is folded into the
//     activation epilogue (avoids per-step accumulator-broadcast movs).
template<int DIN, int U, bool SRC_F16SEQ>
__global__ __launch_bounds__((U / 16) * 32)
void lstm_scan(const float* __restrict__ srcF32,          // layer1 x [B,T,DIN] f32
               const unsigned short* __restrict__ srcH16, // layer>=2 [T,B,DIN] f16
               const float* __restrict__ Wk,               // [DIN,4U] f32
               const float* __restrict__ Wr,               // [U,4U]   f32
               const float* __restrict__ bias,             // [4U]     f32
               unsigned short* __restrict__ outSeq,        // [T,B,U] f16 or null
               float* __restrict__ outLast)                // [B,U] f32 or null
{
  constexpr int K     = DIN + U;        // multiple of 32 for all layers
  constexpr int KT    = K / 32;         // k-tiles of 32
  constexpr int NUT   = U / 16;         // waves per block (one u-tile each)
  constexpr int G4U   = 4 * U;
  constexpr int ABUF  = KT * 512;       // ushorts per A staging buffer
  constexpr int AOFF  = NUT * 4 * KT * 512; // ushort offset of A staging pair

  extern __shared__ unsigned short lds16[];

  const int tid  = threadIdx.x;
  const int nth  = NUT * 32;
  const int wave = tid >> 5;
  const int lane = tid & 31;
  const int n    = lane & 15;
  const int hiL  = lane >> 4;
  const int wgB  = blockIdx.x * 16;     // first batch row of this WG

  // ---- one-time: swizzle concat weights into WMMA B-fragment layout ----
  // B-frag (32x16 f16): (k_local, n) -> lane = n + 16*((k_local>>3)&1),
  //                     elem = (k_local&7) | ((k_local>>4)<<3)
  for (int i = tid; i < K * G4U; i += nth) {
    int col = i % G4U, k = i / G4U;
    float w = (k < DIN) ? Wk[k * G4U + col] : Wr[(k - DIN) * G4U + col];
    int g = col / U, u = col % U;
    int ut = u >> 4, nn = u & 15;
    int kt = k >> 5, kl = k & 31;
    int lB = nn + (((kl >> 3) & 1) << 4);
    int e  = (kl & 7) | ((kl >> 4) << 3);
    lds16[((ut * 4 + g) * KT + kt) * 512 + lB * 16 + e] = f2h(w);
  }
  // zero both A staging buffers (h0 = 0)
  for (int i = tid; i < 2 * ABUF; i += nth) lds16[AOFF + i] = 0;
  __syncthreads();

  // cooperative load of x_t into A staging buffer `par` (A-fragment layout).
  // A-frag (16x32 f16): (m, k_local) -> lane = m + 16*((k_local>>3)&1),
  //                     elem = (k_local&7) | ((k_local>>4)<<3)
  auto load_x = [&](int t, int par) {
    constexpr int NP = 16 * DIN / 2;
    const int base = AOFF + par * ABUF;
    for (int idx = tid; idx < NP; idx += nth) {
      int m = idx / (DIN / 2);
      int k = (idx % (DIN / 2)) * 2;           // even k: (k,k+1) share a lane,
      unsigned int pair;                       // adjacent elems -> one b32 store
      if constexpr (SRC_F16SEQ) {
        const unsigned short* p = srcH16 + ((size_t)t * B_TOT + wgB + m) * DIN + k;
        pair = (unsigned int)p[0] | ((unsigned int)p[1] << 16);
      } else {
        const float* p = srcF32 + ((size_t)(wgB + m) * T_LEN + t) * DIN + k;
        pair = (unsigned int)f2h(p[0]) | ((unsigned int)f2h(p[1]) << 16);
      }
      int kt = k >> 5, kl = k & 31;
      int lA = m + (((kl >> 3) & 1) << 4);
      int e  = (kl & 7) | ((kl >> 4) << 3);
      *(unsigned int*)&lds16[base + kt * 512 + lA * 16 + e] = pair;
    }
  };
  load_x(0, 0);

  // per-lane bias (same column for all 8 accumulator rows)
  float bg[4];
#pragma unroll
  for (int g = 0; g < 4; ++g) bg[g] = bias[g * U + wave * 16 + n];

  float cst[8];
#pragma unroll
  for (int r = 0; r < 8; ++r) cst[r] = 0.f;

  const v8f zacc = {};   // loop-invariant zero C for the peeled first k-tile

  __syncthreads();

  for (int t = 0; t < T_LEN; ++t) {
    const int par = t & 1, nxt = par ^ 1;
    const int aBase = AOFF + par * ABUF;

    v8f acc[4];
    { // peeled kt = 0: C = 0 (inline-zero SRC2; no per-step accumulator init)
      v16h a = *(const v16h*)&lds16[aBase + lane * 16];
#pragma unroll
      for (int g = 0; g < 4; ++g) {
        v16h b = *(const v16h*)&lds16[((wave * 4 + g) * KT) * 512 + lane * 16];
        acc[g] = __builtin_amdgcn_wmma_f32_16x16x32_f16(
            false, a, false, b, (short)0, zacc, false, false);
      }
    }
#pragma unroll
    for (int kt = 1; kt < KT; ++kt) {
      v16h a = *(const v16h*)&lds16[aBase + kt * 512 + lane * 16];
#pragma unroll
      for (int g = 0; g < 4; ++g) {
        v16h b = *(const v16h*)&lds16[((wave * 4 + g) * KT + kt) * 512 + lane * 16];
        acc[g] = __builtin_amdgcn_wmma_f32_16x16x32_f16(
            false, a, false, b, (short)0, acc[g], false, false);
      }
    }

    // gates (bias folded here): i,f,o sigmoid; g relu;
    // c = f*c + i*g; h = o*relu(c)
    float hv[8];
#pragma unroll
    for (int r = 0; r < 8; ++r) {
      float ig = sigmoidf(acc[0][r] + bg[0]);
      float fg = sigmoidf(acc[1][r] + bg[1]);
      float gg = fmaxf(acc[2][r] + bg[2], 0.f);
      float og = sigmoidf(acc[3][r] + bg[3]);
      cst[r] = fg * cst[r] + ig * gg;
      hv[r]  = og * fmaxf(cst[r], 0.f);
    }

    if (outSeq) {
#pragma unroll
      for (int r = 0; r < 8; ++r) {
        int m = r + 8 * hiL;
        outSeq[((size_t)t * B_TOT + wgB + m) * U + wave * 16 + n] = f2h(hv[r]);
      }
    }
    if (outLast && t == T_LEN - 1) {
#pragma unroll
      for (int r = 0; r < 8; ++r) {
        int m = r + 8 * hiL;
        outLast[(size_t)(wgB + m) * U + wave * 16 + n] = hv[r];
      }
    }

    // write h(t+1 input) and x(t+1) into the *other* buffer; the single
    // end-of-step barrier orders these against next step's reads, and the
    // previous barrier ordered them against last step's reads of that buffer.
    {
      int k  = DIN + wave * 16 + n;    // h column position in concat-K
      int kt = k >> 5, kl = k & 31;
      int e  = (kl & 7) | ((kl >> 4) << 3);
      const int hBase = AOFF + nxt * ABUF + kt * 512;
#pragma unroll
      for (int r = 0; r < 8; ++r) {
        int m  = r + 8 * hiL;
        int lA = m + (((kl >> 3) & 1) << 4);
        lds16[hBase + lA * 16 + e] = f2h(hv[r]);
      }
    }
    if (t + 1 < T_LEN) load_x(t + 1, nxt);
    __syncthreads();
  }
}

// tiny dense head: [B,32] @ [32,18] + bd
__global__ void dense_kernel(const float* __restrict__ h3,
                             const float* __restrict__ Wd,
                             const float* __restrict__ bd,
                             float* __restrict__ out)
{
  int i = blockIdx.x * blockDim.x + threadIdx.x;
  if (i >= B_TOT * 18) return;
  int b = i / 18, o = i % 18;
  float s = bd[o];
#pragma unroll
  for (int u = 0; u < 32; ++u) s += h3[b * 32 + u] * Wd[u * 18 + o];
  out[i] = s;
}

extern "C" void kernel_launch(void* const* d_in, const int* in_sizes, int n_in,
                              void* d_out, int out_size, void* d_ws, size_t ws_size,
                              hipStream_t stream) {
  const float* x   = (const float*)d_in[0];
  const float* Wk1 = (const float*)d_in[1];
  const float* Wr1 = (const float*)d_in[2];
  const float* b1  = (const float*)d_in[3];
  const float* Wk2 = (const float*)d_in[4];
  const float* Wr2 = (const float*)d_in[5];
  const float* b2  = (const float*)d_in[6];
  const float* Wk3 = (const float*)d_in[7];
  const float* Wr3 = (const float*)d_in[8];
  const float* b3  = (const float*)d_in[9];
  const float* Wd  = (const float*)d_in[10];
  const float* bd  = (const float*)d_in[11];

  char* ws = (char*)d_ws;
  size_t h1_bytes = (size_t)T_LEN * B_TOT * 128 * 2; // f16 seq, layer1 out
  size_t h2_bytes = (size_t)T_LEN * B_TOT * 64 * 2;  // f16 seq, layer2 out
  unsigned short* h1 = (unsigned short*)ws;
  unsigned short* h2 = (unsigned short*)(ws + h1_bytes);
  float*          h3 = (float*)(ws + h1_bytes + h2_bytes); // f32 [B,32]

  // LDS bytes: (NUT*4*KT + 2*KT) fragments * 1024 B  (weights + double A-buf)
  int lds1 = (8 * 4 * 6 + 2 * 6) * 1024; // 208896 (<= 320KB WGP LDS)
  int lds2 = (4 * 4 * 6 + 2 * 6) * 1024; // 110592
  int lds3 = (2 * 4 * 3 + 2 * 3) * 1024; //  30720
  (void)hipFuncSetAttribute(
      reinterpret_cast<const void*>(&lstm_scan<64, 128, false>),
      hipFuncAttributeMaxDynamicSharedMemorySize, lds1);
  (void)hipFuncSetAttribute(
      reinterpret_cast<const void*>(&lstm_scan<128, 64, true>),
      hipFuncAttributeMaxDynamicSharedMemorySize, lds2);
  (void)hipFuncSetAttribute(
      reinterpret_cast<const void*>(&lstm_scan<64, 32, true>),
      hipFuncAttributeMaxDynamicSharedMemorySize, lds3);

  lstm_scan<64, 128, false><<<16, 256, lds1, stream>>>(
      x, nullptr, Wk1, Wr1, b1, h1, nullptr);
  lstm_scan<128, 64, true><<<16, 128, lds2, stream>>>(
      nullptr, h1, Wk2, Wr2, b2, h2, nullptr);
  lstm_scan<64, 32, true><<<16, 64, lds3, stream>>>(
      nullptr, h2, Wk3, Wr3, b3, nullptr, h3);
  dense_kernel<<<(B_TOT * 18 + 127) / 128, 128, 0, stream>>>(
      h3, Wd, bd, (float*)d_out);
}
